// AblationCAM_7138235645960
// MI455X (gfx1250) — matrix operand order repaired and checked by reading, not compiled, not run on hardware
//
#include <hip/hip_runtime.h>
#include <hip/hip_bf16.h>
#include <math.h>

// ---------------------------------------------------------------------------
// Types for CDNA5 WMMA
// ---------------------------------------------------------------------------
typedef __attribute__((ext_vector_type(16))) __bf16 v16bf;
typedef __attribute__((ext_vector_type(8)))  float  v8f;

__device__ __forceinline__ __bf16 f2bf(float f) {
  // native fp32 -> bf16 conversion (RNE); clang emits v_cvt_pk_bf16_f32
  return (__bf16)f;
}

__device__ __forceinline__ float waveReduceSum(float v) {
  #pragma unroll
  for (int m = 16; m > 0; m >>= 1) v += __shfl_xor(v, m, 32);
  return v;
}

// gfx1250 async global->LDS DMA (GLOBAL_LOAD_ASYNC_TO_LDS_B128, ASYNCcnt).
// LDS dest is the per-lane 32-bit LDS offset (low half of the generic pointer:
// shared aperture occupies ADDR[63:32]).
__device__ __forceinline__ void asyncCopyB128(const void* g, void* l) {
  asm volatile("global_load_async_to_lds_b128 %0, %1, off"
               :: "v"((unsigned)(size_t)l), "v"((unsigned long long)(size_t)g)
               : "memory");
}
__device__ __forceinline__ void waitAsync() {
  asm volatile("s_wait_asynccnt 0" ::: "memory");
}

// ---------------------------------------------------------------------------
// K1: 2x2 maxpool  [512,14,14] -> mp[25088] (c-major, 7x7 per channel)
// ---------------------------------------------------------------------------
__global__ void k_maxpool(const float* __restrict__ act, float* __restrict__ mp) {
  int f = blockIdx.x * blockDim.x + threadIdx.x;
  if (f >= 25088) return;
  int c = f / 49, p = f % 49;
  int y = (p / 7) * 2, x = (p % 7) * 2;
  const float* b = act + c * 196 + y * 14 + x;
  mp[f] = fmaxf(fmaxf(b[0], b[1]), fmaxf(b[14], b[15]));
}

// ---------------------------------------------------------------------------
// K2: block partials P[c][j] = dot(W1[j, c*49 : c*49+49], mp[c*49 : ...])
// One block per output row j; thread t covers channels 2t, 2t+1 = 98
// contiguous floats (392 B, 8-aligned -> float2). Reads W1 exactly once
// (411 MB) -> this stream is the whole pipeline's bandwidth floor.
// ---------------------------------------------------------------------------
__global__ void __launch_bounds__(256) k_partials(const float* __restrict__ W1,
                                                  const float* __restrict__ mp,
                                                  float* __restrict__ P) {
  int j = blockIdx.x;            // 0..4095
  int t = threadIdx.x;           // 0..255
  const float* r0 = W1 + (size_t)j * 25088 + t * 98;
  const float* m0 = mp + t * 98;
  const float2* r2 = (const float2*)r0;
  const float2* m2 = (const float2*)m0;
  float s0 = 0.f, s1 = 0.f;
  #pragma unroll
  for (int i = 0; i < 24; ++i) {        // floats 0..47 of channel 2t
    float2 a = r2[i], b = m2[i];
    s0 += a.x * b.x + a.y * b.y;
  }
  s0 += r0[48] * m0[48];                // float 48
  s1 += r0[49] * m0[49];                // float 0 of channel 2t+1
  const float2* r2b = (const float2*)(r0 + 50);
  const float2* m2b = (const float2*)(m0 + 50);
  #pragma unroll
  for (int i = 0; i < 24; ++i) {        // floats 1..48 of channel 2t+1
    float2 a = r2b[i], b = m2b[i];
    s1 += a.x * b.x + a.y * b.y;
  }
  P[(size_t)(2 * t) * 4096 + j]     = s0;
  P[(size_t)(2 * t + 1) * 4096 + j] = s1;
}

// ---------------------------------------------------------------------------
// K3: h1_pre_base[j] = b1[j] + sum_c P[c][j];   h1_base = relu(h1_pre_base)
// ---------------------------------------------------------------------------
__global__ void k_colsum(const float* __restrict__ P, const float* __restrict__ b1,
                         float* __restrict__ h1pre, float* __restrict__ h1base) {
  int j = blockIdx.x * 256 + threadIdx.x;   // < 4096
  float s = b1[j];
  for (int c = 0; c < 512; ++c) s += P[(size_t)c * 4096 + j];
  h1pre[j] = s;
  h1base[j] = fmaxf(s, 0.f);
}

// ---------------------------------------------------------------------------
// K4/K5: GEMV  y = (relu?)(M @ x + bias), one wave per output row
// ---------------------------------------------------------------------------
__global__ void __launch_bounds__(256) k_gemv(const float* __restrict__ M,
                                              const float* __restrict__ x,
                                              const float* __restrict__ bias,
                                              float* __restrict__ y,
                                              int K, int relu) {
  int w = threadIdx.x >> 5, lane = threadIdx.x & 31;
  int row = blockIdx.x * 8 + w;
  const float* r = M + (size_t)row * K;
  float s = 0.f;
  for (int k = lane; k < K; k += 32) s += r[k] * x[k];
  s = waveReduceSum(s);
  if (lane == 0) {
    s += bias[row];
    y[row] = relu ? fmaxf(s, 0.f) : s;
  }
}

// ---------------------------------------------------------------------------
// K6: argmax of logits[1000] -> sc[0]=class(int), sc[1]=base; also writes
//     d_out[50176] = (float)class (second tuple element of the reference)
// ---------------------------------------------------------------------------
__global__ void k_argmax(const float* __restrict__ logits, float* __restrict__ sc,
                         float* __restrict__ out) {
  __shared__ float sv[1024];
  __shared__ int   si[1024];
  int t = threadIdx.x;
  sv[t] = (t < 1000) ? logits[t] : -3.4e38f;
  si[t] = t;
  __syncthreads();
  for (int s = 512; s > 0; s >>= 1) {
    if (t < s) {
      float v2 = sv[t + s]; int i2 = si[t + s];
      if (v2 > sv[t] || (v2 == sv[t] && i2 < si[t])) { sv[t] = v2; si[t] = i2; }
    }
    __syncthreads();
  }
  if (t == 0) {
    ((int*)sc)[0] = si[0];
    sc[1] = sv[0];
    out[50176] = (float)si[0];
  }
}

// ---------------------------------------------------------------------------
// K7: H1[c][j] = bf16(relu(h1_pre_base[j] - P[c][j]))   [512,4096] bf16
//     float4 in, packed 4xbf16 (uint2) out.
// ---------------------------------------------------------------------------
__global__ void k_buildH1(const float* __restrict__ h1pre, const float* __restrict__ P,
                          __bf16* __restrict__ H1) {
  int q = blockIdx.x * 256 + threadIdx.x;   // < 524288 quads
  int idx = q * 4;
  int j = idx & 4095;
  float4 p = *(const float4*)(P + idx);
  float4 h = *(const float4*)(h1pre + j);
  union { __bf16 h4[4]; uint2 u; } o;
  o.h4[0] = f2bf(fmaxf(h.x - p.x, 0.f));
  o.h4[1] = f2bf(fmaxf(h.y - p.y, 0.f));
  o.h4[2] = f2bf(fmaxf(h.z - p.z, 0.f));
  o.h4[3] = f2bf(fmaxf(h.w - p.w, 0.f));
  *(uint2*)(H1 + idx) = o.u;
}

// ---------------------------------------------------------------------------
// K8: WMMA GEMM  H2 = relu(H1 @ W2^T + b2)
//   A = H1 [M=512, K=4096] bf16 row-major; B[k][n] = W2[n][k] (bf16 on the fly)
//   Block tile 128(M) x 128(N), K-step 32, 8 waves as 4(wm) x 2(wn), each
//   wave computes 32x64 = 2x4 tiles of v_wmma_f32_16x16x32_bf16.
//   A tile: GLOBAL_LOAD_ASYNC_TO_LDS_B128 (no VGPR staging -> kills the
//   scratch spill of the staged quads seen in earlier builds; ASYNCcnt fenced
//   with s_wait_asynccnt 0 before the barrier). B tile: register-staged fp32
//   loads converted via v_cvt_pk_bf16_f32 into LDS. Both LDS operands are
//   lane-contiguous so every fragment is 2x ds_load_b128. Double-buffered,
//   one barrier per K-step.
//   Fragment lane mapping per ISA 7.12.2 (16-bit A 16x32 / B 32x16, f32 C/D):
//   A: lane<16: row=lane, K = {0..7,16..23}; lane>=16: row=lane-16, K={8..15,24..31}
//   B: lane<16: col=lane, K = {0..7,16..23}; lane>=16: col=lane-16, K={8..15,24..31}
//   D: vgpr i holds M = i + 8*(lane>=16), N = lane&15
// ---------------------------------------------------------------------------
__global__ void __launch_bounds__(256, 1) k_gemm_wmma(const __bf16* __restrict__ A,
                                                      const float* __restrict__ W2,
                                                      const float* __restrict__ b2,
                                                      float* __restrict__ D) {
  __shared__ __bf16 As[2][128][40];   // [M][K] + pad
  __shared__ __bf16 Bs[2][128][40];   // [N][K] + pad
  const int t    = threadIdx.x;
  const int bm   = blockIdx.y * 128;
  const int bn   = blockIdx.x * 128;
  const int w    = t >> 5, lane = t & 31;
  const int wm   = w >> 1, wn = w & 1;          // 4 x 2 waves
  const int half = lane >> 4, r16 = lane & 15;

  v8f acc[2][4];
  #pragma unroll
  for (int ms = 0; ms < 2; ++ms)
    #pragma unroll
    for (int ns = 0; ns < 4; ++ns)
      #pragma unroll
      for (int i = 0; i < 8; ++i) acc[ms][ns][i] = 0.f;

  float4 bReg[4];

  auto loadA_async = [&](int kb, int s) {       // A: 4096 bf16, 2 x b128/thread
    #pragma unroll
    for (int e = 0; e < 2; ++e) {
      int li = t + e * 256, r = li >> 2, kq = (li & 3) * 8;
      asyncCopyB128(A + (size_t)(bm + r) * 4096 + kb + kq, &As[s][r][kq]);
    }
  };
  auto loadB = [&](int kb) {                    // B: 4096 f32, 16/thread
    #pragma unroll
    for (int e = 0; e < 4; ++e) {
      int li = t + e * 256, n = li >> 3, kq = (li & 7) * 4;
      bReg[e] = *(const float4*)(W2 + (size_t)(bn + n) * 4096 + kb + kq);
    }
  };
  auto storeB = [&](int s) {
    #pragma unroll
    for (int e = 0; e < 4; ++e) {
      int li = t + e * 256, n = li >> 3, kq = (li & 7) * 4;
      union { __bf16 h4[4]; uint2 u; } o;
      o.h4[0] = f2bf(bReg[e].x);
      o.h4[1] = f2bf(bReg[e].y);
      o.h4[2] = f2bf(bReg[e].z);
      o.h4[3] = f2bf(bReg[e].w);
      *(uint2*)&Bs[s][n][kq] = o.u;
    }
  };
  auto compute = [&](int s) {
    v16bf afrag[2];
    #pragma unroll
    for (int ms = 0; ms < 2; ++ms) {
      const int arow = wm * 32 + ms * 16 + r16;
      union { uint4 q[2]; v16bf v; } ua;
      ua.q[0] = *(const uint4*)&As[s][arow][half * 8];
      ua.q[1] = *(const uint4*)&As[s][arow][16 + half * 8];
      afrag[ms] = ua.v;
    }
    #pragma unroll
    for (int ns = 0; ns < 4; ++ns) {
      const int col = wn * 64 + ns * 16 + r16;
      union { uint4 q[2]; v16bf v; } ub;
      ub.q[0] = *(const uint4*)&Bs[s][col][half * 8];
      ub.q[1] = *(const uint4*)&Bs[s][col][16 + half * 8];
      #pragma unroll
      for (int ms = 0; ms < 2; ++ms)
        acc[ms][ns] = __builtin_amdgcn_wmma_f32_16x16x32_bf16(
            false, afrag[ms], false, ub.v, (short)0, acc[ms][ns], false, false);
    }
  };

  loadA_async(0, 0);
  loadB(0);
  storeB(0);
  waitAsync();
  __syncthreads();
  for (int ki = 0; ki < 128; ++ki) {
    const int cur = ki & 1;
    if (ki < 127) {
      loadA_async((ki + 1) * 32, cur ^ 1);      // DMA straight into LDS
      loadB((ki + 1) * 32);                     // staged (needs cvt)
    }
    compute(cur);
    if (ki < 127) storeB(cur ^ 1);
    waitAsync();                                // async A writes visible
    __syncthreads();                            // before anyone reads them
  }

  #pragma unroll
  for (int ms = 0; ms < 2; ++ms)
    #pragma unroll
    for (int ns = 0; ns < 4; ++ns) {
      const int col = bn + wn * 64 + ns * 16 + r16;
      const float bias = b2[col];
      #pragma unroll
      for (int i = 0; i < 8; ++i) {
        const int row = bm + wm * 32 + ms * 16 + half * 8 + i;
        float v = acc[ms][ns][i] + bias;
        D[(size_t)row * 4096 + col] = v > 0.f ? v : 0.f;
      }
    }
}

// ---------------------------------------------------------------------------
// K9: alpha[c] = (base - (H2[c]·W3[class] + b3[class])) / base
//     (also re-initializes the min/max cells every call)
// ---------------------------------------------------------------------------
__global__ void __launch_bounds__(256) k_alpha(const float* __restrict__ H2,
                                               const float* __restrict__ W3,
                                               const float* __restrict__ b3,
                                               const float* __restrict__ sc,
                                               float* __restrict__ alpha,
                                               unsigned* __restrict__ mm) {
  if (blockIdx.x == 0 && threadIdx.x == 0) { mm[0] = 0x7f7fffffu; mm[1] = 0u; }
  int w = threadIdx.x >> 5, lane = threadIdx.x & 31;
  int c = blockIdx.x * 8 + w;   // < 512
  int cls = ((const int*)sc)[0];
  float base = sc[1];
  const float* h  = H2 + (size_t)c * 4096;
  const float* wr = W3 + (size_t)cls * 4096;
  float s = 0.f;
  for (int k = lane; k < 4096; k += 32) s += h[k] * wr[k];
  s = waveReduceSum(s);
  if (lane == 0) alpha[c] = (base - (s + b3[cls])) / base;
}

// ---------------------------------------------------------------------------
// K10: weighted[p] = sum_c alpha[c] * act[c][p]   (p over 14x14)
// ---------------------------------------------------------------------------
__global__ void k_weighted(const float* __restrict__ act, const float* __restrict__ alpha,
                           float* __restrict__ wmap) {
  int p = threadIdx.x;
  if (p >= 196) return;
  float s = 0.f;
  for (int c = 0; c < 512; ++c) s += alpha[c] * act[c * 196 + p];
  wmap[p] = s;
}

// ---------------------------------------------------------------------------
// K11: bilinear 14->224 (half-pixel centers, clamped taps == jax renormalized
//      triangle at edges), relu, global min/max via monotone-uint atomics
// ---------------------------------------------------------------------------
__global__ void __launch_bounds__(256) k_resize(const float* __restrict__ wmap,
                                                float* __restrict__ raw,
                                                unsigned* __restrict__ mm) {
  int o = blockIdx.x * 256 + threadIdx.x;   // < 50176
  int oy = o / 224, ox = o % 224;
  float fy = (oy + 0.5f) * (14.f / 224.f) - 0.5f;
  float fx = (ox + 0.5f) * (14.f / 224.f) - 0.5f;
  int y0 = (int)floorf(fy); float wy = fy - (float)y0;
  int x0 = (int)floorf(fx); float wx = fx - (float)x0;
  int y0c = min(max(y0, 0), 13), y1c = min(max(y0 + 1, 0), 13);
  int x0c = min(max(x0, 0), 13), x1c = min(max(x0 + 1, 0), 13);
  float v00 = wmap[y0c * 14 + x0c], v01 = wmap[y0c * 14 + x1c];
  float v10 = wmap[y1c * 14 + x0c], v11 = wmap[y1c * 14 + x1c];
  float v = (1.f - wy) * ((1.f - wx) * v00 + wx * v01)
          +        wy  * ((1.f - wx) * v10 + wx * v11);
  v = fmaxf(v, 0.f);
  raw[o] = v;
  float mx = v, mn = v;
  #pragma unroll
  for (int m = 16; m > 0; m >>= 1) {
    mx = fmaxf(mx, __shfl_xor(mx, m, 32));
    mn = fminf(mn, __shfl_xor(mn, m, 32));
  }
  if ((threadIdx.x & 31) == 0) {   // values >= 0 -> uint order == float order
    atomicMax(&mm[1], __float_as_uint(mx));
    atomicMin(&mm[0], __float_as_uint(mn));
  }
}

// ---------------------------------------------------------------------------
// K12: min-max normalize into d_out[0..50175]
// ---------------------------------------------------------------------------
__global__ void k_norm(const float* __restrict__ raw, const unsigned* __restrict__ mm,
                       float* __restrict__ out) {
  int o = blockIdx.x * 256 + threadIdx.x;
  float mn = __uint_as_float(mm[0]), mx = __uint_as_float(mm[1]);
  float v = raw[o];
  out[o] = (mx != mn) ? (v - mn) / (mx - mn) : v;
}

// ---------------------------------------------------------------------------
// Host launcher
// ---------------------------------------------------------------------------
extern "C" void kernel_launch(void* const* d_in, const int* in_sizes, int n_in,
                              void* d_out, int out_size, void* d_ws, size_t ws_size,
                              hipStream_t stream) {
  (void)in_sizes; (void)n_in; (void)out_size; (void)ws_size;
  const float* act = (const float*)d_in[0];   // [1,512,14,14]
  const float* W1  = (const float*)d_in[1];   // [4096,25088]
  const float* b1  = (const float*)d_in[2];
  const float* W2  = (const float*)d_in[3];   // [4096,4096]
  const float* b2  = (const float*)d_in[4];
  const float* W3  = (const float*)d_in[5];   // [1000,4096]
  const float* b3  = (const float*)d_in[6];
  float* out = (float*)d_out;                 // [224*224 + 1]

  // workspace layout (float element offsets)
  float* F        = (float*)d_ws;
  float* mp       = F;                         // 25088
  float* P        = F + 32768;                 // 512*4096
  float* h1pre    = F + 32768 + 2097152;       // 4096
  float* h1base   = h1pre + 4096;              // 4096
  float* h2base   = h1base + 4096;             // 4096
  float* logits   = h2base + 4096;             // 1000 (pad 1024)
  float* alpha    = logits + 1024;             // 512
  float* wmap     = alpha + 512;               // 196 (pad 256)
  float* raw      = wmap + 256;                // 50176
  float* sc       = raw + 50176;               // [0]=class bits, [1]=base
  unsigned* mm    = (unsigned*)(sc + 2);       // [0]=min bits, [1]=max bits
  float* H2       = sc + 16;                   // 512*4096 f32
  __bf16* H1      = (__bf16*)(H2 + 2097152);   // 512*4096 bf16

  k_maxpool <<<98,   256, 0, stream>>>(act, mp);
  k_partials<<<4096, 256, 0, stream>>>(W1, mp, P);
  k_colsum  <<<16,   256, 0, stream>>>(P, b1, h1pre, h1base);
  k_gemv    <<<512,  256, 0, stream>>>(W2, h1base, b2, h2base, 4096, 1);
  k_gemv    <<<125,  256, 0, stream>>>(W3, h2base, b3, logits, 4096, 0);
  k_argmax  <<<1,   1024, 0, stream>>>(logits, sc, out);
  k_buildH1 <<<2048, 256, 0, stream>>>(h1pre, P, H1);
  k_gemm_wmma<<<dim3(32, 4), 256, 0, stream>>>(H1, W2, b2, H2);
  k_alpha   <<<64,   256, 0, stream>>>(H2, W3, b3, sc, alpha, mm);
  k_weighted<<<1,    224, 0, stream>>>(act, alpha, wmap);
  k_resize  <<<196,  256, 0, stream>>>(wmap, raw, mm);
  k_norm    <<<196,  256, 0, stream>>>(raw, mm, out);
}
